// GATemb_17291538334380
// MI455X (gfx1250) — compile-verified, matrix operand order checked by reading, not run on hardware
//
#include <hip/hip_runtime.h>

// ---------------- problem constants (from reference) ----------------
#define N_NODES 50000
#define E_EDGES 800000
#define ET      (E_EDGES + N_NODES)   // edges + self loops
#define IN_CH   128
#define HID     64
#define HEADS   3
#define HC      (HEADS * HID)         // 192
#define F1      64
#define OUT_CH  32
#define SLOPE   0.2f

typedef __bf16 bf16;
typedef __attribute__((ext_vector_type(16))) __bf16 bf16x16;
typedef __attribute__((ext_vector_type(8)))  __bf16 bf16x8;
typedef __attribute__((ext_vector_type(8)))  float  f32x8;

// ---------------- small helpers ----------------
__device__ __forceinline__ unsigned f2ord(float f) {
    unsigned u = __float_as_uint(f);
    return (u & 0x80000000u) ? ~u : (u | 0x80000000u);
}
__device__ __forceinline__ float ord2f(unsigned u) {
    return (u & 0x80000000u) ? __uint_as_float(u & 0x7FFFFFFFu)
                             : __uint_as_float(~u);
}
__device__ __forceinline__ void edge_sd(const int* __restrict__ ei, int e, int& s, int& d) {
    if (e < E_EDGES) { s = ei[e]; d = ei[E_EDGES + e]; }
    else             { s = e - E_EDGES; d = s; }
}

// ---------------- conversion kernels ----------------
__global__ void gat_cvt_bf16(const float* __restrict__ x, bf16* __restrict__ y, long n) {
    long i = (long)blockIdx.x * blockDim.x + threadIdx.x;
    long stride = (long)gridDim.x * blockDim.x;
    for (; i < n; i += stride) y[i] = (bf16)x[i];
}

// W[K][Nout] fp32 row-major  ->  Wt[Nout][K] bf16 (so B-frag loads are contiguous)
__global__ void gat_wt_transpose(const float* __restrict__ W, bf16* __restrict__ Wt,
                                 int K, int Nout) {
    int t = blockIdx.x * blockDim.x + threadIdx.x;
    if (t >= K * Nout) return;
    int k = t / Nout, c = t % Nout;
    Wt[(size_t)c * K + k] = (bf16)W[t];
}

__global__ void gat_zero_f32(float* __restrict__ p, long n) {
    long i = (long)blockIdx.x * blockDim.x + threadIdx.x;
    long stride = (long)gridDim.x * blockDim.x;
    for (; i < n; i += stride) p[i] = 0.0f;
}

// ---------------- WMMA bf16 GEMM: C[M,Nout] = A[M,K] @ Bt[Nout,K]^T ----------------
// A row-major bf16, Bt = column-major weights (i.e. W transposed), fp32 accum.
// 4 waves per block, one 16x16 tile per wave.
__global__ __launch_bounds__(128) void gat_wmma_gemm(
    const bf16* __restrict__ A, const bf16* __restrict__ Bt,
    const float* __restrict__ bias, float* __restrict__ C,
    bf16* __restrict__ Cbf, int M, int K, int Nout)
{
    const int lane = threadIdx.x & 31;
    const int rowTile = blockIdx.x << 4;
    const int colTileIdx = blockIdx.y * blockDim.y + threadIdx.y;
    if ((colTileIdx << 4) >= Nout) return;
    const int colTile = colTileIdx << 4;

    const int half = lane >> 4;    // 0: K {0..7,16..23}; 1: K {8..15,24..31}
    const int l15  = lane & 15;
    const bf16* Arow = A  + (size_t)(rowTile + l15) * K + half * 8;
    const bf16* Bcol = Bt + (size_t)(colTile + l15) * K + half * 8;

    f32x8 acc = {};
    for (int k0 = 0; k0 < K; k0 += 32) {
        if (k0 + 32 < K) {
            __builtin_prefetch(Arow + k0 + 32, 0, 0);   // global_prefetch_b8
            __builtin_prefetch(Bcol + k0 + 32, 0, 0);
        }
        bf16x8 a_lo = *(const bf16x8*)(Arow + k0);
        bf16x8 a_hi = *(const bf16x8*)(Arow + k0 + 16);
        bf16x8 b_lo = *(const bf16x8*)(Bcol + k0);
        bf16x8 b_hi = *(const bf16x8*)(Bcol + k0 + 16);
        bf16x16 af = __builtin_shufflevector(a_lo, a_hi,
                        0,1,2,3,4,5,6,7,8,9,10,11,12,13,14,15);
        bf16x16 bff = __builtin_shufflevector(b_lo, b_hi,
                        0,1,2,3,4,5,6,7,8,9,10,11,12,13,14,15);
        acc = __builtin_amdgcn_wmma_f32_16x16x32_bf16(
                  false, af, false, bff, (short)0, acc, false, false);
    }

    const int col = colTile + l15;              // one column per lane in C/D
    const int rbase = rowTile + half * 8;       // lanes 16-31 hold M=8..15
    const float bv = bias ? bias[col] : 0.0f;
#pragma unroll
    for (int v = 0; v < 8; ++v) {
        float val = acc[v] + bv;
        size_t idx = (size_t)(rbase + v) * Nout + col;
        if (C)   C[idx]   = val;
        if (Cbf) Cbf[idx] = (bf16)val;
    }
}

// ---------------- attention score projections: as/ad [N, HEADS] ----------------
__global__ void gat_alpha(const float* __restrict__ h,
                          const float* __restrict__ a_src,
                          const float* __restrict__ a_dst,
                          float* __restrict__ as_, float* __restrict__ ad_)
{
    int t = blockIdx.x * blockDim.x + threadIdx.x;
    if (t >= N_NODES * HEADS) return;
    int n = t / HEADS, hd = t % HEADS;
    const float* hp = h + (size_t)n * HC + hd * HID;
    const float* s  = a_src + hd * HID;
    const float* d  = a_dst + hd * HID;
    float ss = 0.f, dd = 0.f;
#pragma unroll 8
    for (int c = 0; c < HID; ++c) { float v = hp[c]; ss += v * s[c]; dd += v * d[c]; }
    as_[t] = ss; ad_[t] = dd;
}

// ---------------- segment softmax pass 1: per-dst max ----------------
__global__ void gat_edge_max(const int* __restrict__ ei,
                             const float* __restrict__ as_, const float* __restrict__ ad_,
                             unsigned* __restrict__ mx)
{
    int e = blockIdx.x * blockDim.x + threadIdx.x;
    if (e >= ET) return;
    int s, d; edge_sd(ei, e, s, d);
#pragma unroll
    for (int hd = 0; hd < HEADS; ++hd) {
        float sc = as_[s * HEADS + hd] + ad_[d * HEADS + hd];
        sc = sc > 0.f ? sc : SLOPE * sc;                 // leaky_relu
        atomicMax(&mx[d * HEADS + hd], f2ord(sc));
    }
}

// ---------------- pass 2: per-dst sum of exp ----------------
__global__ void gat_edge_sum(const int* __restrict__ ei,
                             const float* __restrict__ as_, const float* __restrict__ ad_,
                             const unsigned* __restrict__ mx, float* __restrict__ dn)
{
    int e = blockIdx.x * blockDim.x + threadIdx.x;
    if (e >= ET) return;
    int s, d; edge_sd(ei, e, s, d);
#pragma unroll
    for (int hd = 0; hd < HEADS; ++hd) {
        float sc = as_[s * HEADS + hd] + ad_[d * HEADS + hd];
        sc = sc > 0.f ? sc : SLOPE * sc;
        float m = ord2f(mx[d * HEADS + hd]);
        atomicAdd(&dn[d * HEADS + hd], __expf(sc - m));
    }
}

// ---------------- pass 3: weighted scatter of messages ----------------
__global__ void gat_edge_msg(const int* __restrict__ ei,
                             const float* __restrict__ as_, const float* __restrict__ ad_,
                             const unsigned* __restrict__ mx, const float* __restrict__ dn,
                             const float* __restrict__ h, float* __restrict__ agg)
{
    int t = blockIdx.x * blockDim.x + threadIdx.x;
    if (t >= ET * HEADS) return;
    int e = t / HEADS, hd = t % HEADS;
    int s, d; edge_sd(ei, e, s, d);
    float sc = as_[s * HEADS + hd] + ad_[d * HEADS + hd];
    sc = sc > 0.f ? sc : SLOPE * sc;
    float m = ord2f(mx[d * HEADS + hd]);
    float alpha = __expf(sc - m) / (dn[d * HEADS + hd] + 1e-16f);

    const float4* hp = (const float4*)(h + (size_t)s * HC + hd * HID);
    float* op = agg + (size_t)d * HC + hd * HID;
    __builtin_prefetch(hp, 0, 0);
#pragma unroll
    for (int c4 = 0; c4 < HID / 4; ++c4) {
        float4 v = hp[c4];
        atomicAdd(op + c4 * 4 + 0, v.x * alpha);
        atomicAdd(op + c4 * 4 + 1, v.y * alpha);
        atomicAdd(op + c4 * 4 + 2, v.z * alpha);
        atomicAdd(op + c4 * 4 + 3, v.w * alpha);
    }
}

// ---------------- epilogue: +bias, ELU, to bf16 for next GEMM ----------------
__global__ void gat_post_elu_bf16(const float* __restrict__ agg,
                                  const float* __restrict__ b, bf16* __restrict__ outb)
{
    long i = (long)blockIdx.x * blockDim.x + threadIdx.x;
    long n = (long)N_NODES * HC;
    long stride = (long)gridDim.x * blockDim.x;
    for (; i < n; i += stride) {
        float v = agg[i] + b[(int)(i % HC)];
        v = v > 0.f ? v : (__expf(v) - 1.f);   // elu
        outb[i] = (bf16)v;
    }
}

// ---------------- host driver ----------------
extern "C" void kernel_launch(void* const* d_in, const int* in_sizes, int n_in,
                              void* d_out, int out_size, void* d_ws, size_t ws_size,
                              hipStream_t stream) {
    const float* x       = (const float*)d_in[0];
    const int*   ei      = (const int*)  d_in[1];
    const float* W1      = (const float*)d_in[2];
    const float* a_src1  = (const float*)d_in[3];
    const float* a_dst1  = (const float*)d_in[4];
    const float* b1      = (const float*)d_in[5];
    const float* W2      = (const float*)d_in[6];
    const float* a_src2  = (const float*)d_in[7];
    const float* a_dst2  = (const float*)d_in[8];
    const float* b2      = (const float*)d_in[9];
    const float* Wf1     = (const float*)d_in[10];
    const float* bf1     = (const float*)d_in[11];
    const float* Wf2     = (const float*)d_in[12];
    const float* bf2     = (const float*)d_in[13];
    (void)in_sizes; (void)n_in; (void)out_size; (void)ws_size;

    // workspace carve-out (256B aligned)
    size_t off = 0;
    auto carve = [&](size_t bytes) -> void* {
        void* p = (char*)d_ws + off;
        off += (bytes + 255) & ~(size_t)255;
        return p;
    };
    bf16*     bufXb = (bf16*)    carve((size_t)N_NODES * HC * sizeof(bf16));   // GEMM input (bf16)
    float*    bufH  = (float*)   carve((size_t)N_NODES * HC * sizeof(float));  // GEMM output h
    float*    agg   = (float*)   carve((size_t)N_NODES * HC * sizeof(float));  // scatter accum
    bf16*     f1b   = (bf16*)    carve((size_t)N_NODES * F1 * sizeof(bf16));   // fc1 bf16 out
    bf16*     W1t   = (bf16*)    carve((size_t)HC * IN_CH * sizeof(bf16));
    bf16*     W2t   = (bf16*)    carve((size_t)HC * HC * sizeof(bf16));
    bf16*     Wf1t  = (bf16*)    carve((size_t)F1 * HC * sizeof(bf16));
    bf16*     Wf2t  = (bf16*)    carve((size_t)OUT_CH * F1 * sizeof(bf16));
    float*    as_   = (float*)   carve((size_t)N_NODES * HEADS * sizeof(float));
    float*    ad_   = (float*)   carve((size_t)N_NODES * HEADS * sizeof(float));
    unsigned* mx    = (unsigned*)carve((size_t)N_NODES * HEADS * sizeof(unsigned));
    float*    dn    = (float*)   carve((size_t)N_NODES * HEADS * sizeof(float));

    const int TB = 256;
    auto cdiv = [](long a, long b) { return (int)((a + b - 1) / b); };

    // --- prep: bf16 x, transposed bf16 weights ---
    gat_cvt_bf16<<<2048, TB, 0, stream>>>(x, bufXb, (long)N_NODES * IN_CH);
    gat_wt_transpose<<<cdiv((long)IN_CH * HC, TB), TB, 0, stream>>>(W1,  W1t,  IN_CH, HC);
    gat_wt_transpose<<<cdiv((long)HC * HC,   TB), TB, 0, stream>>>(W2,  W2t,  HC,    HC);
    gat_wt_transpose<<<cdiv((long)HC * F1,   TB), TB, 0, stream>>>(Wf1, Wf1t, HC,    F1);
    gat_wt_transpose<<<cdiv((long)F1 * OUT_CH, TB), TB, 0, stream>>>(Wf2, Wf2t, F1, OUT_CH);

    dim3 gblock(32, 4);
    auto gemm = [&](const bf16* A, const bf16* Bt, const float* bias,
                    float* C, bf16* Cbf, int K, int Nout) {
        dim3 grid(N_NODES / 16, ((Nout / 16) + 3) / 4);
        gat_wmma_gemm<<<grid, gblock, 0, stream>>>(A, Bt, bias, C, Cbf, N_NODES, K, Nout);
    };

    auto gat_layer = [&](const bf16* in_bf, const bf16* Wt, int Kin,
                         const float* a_s, const float* a_d, const float* bias,
                         bf16* out_bf) {
        gemm(in_bf, Wt, nullptr, bufH, nullptr, Kin, HC);              // h = x @ W  (WMMA)
        gat_alpha<<<cdiv((long)N_NODES * HEADS, TB), TB, 0, stream>>>(bufH, a_s, a_d, as_, ad_);
        gat_zero_f32<<<2048, TB, 0, stream>>>(agg, (long)N_NODES * HC);
        gat_zero_f32<<<64,   TB, 0, stream>>>(dn,  (long)N_NODES * HEADS);
        gat_zero_f32<<<64,   TB, 0, stream>>>((float*)mx, (long)N_NODES * HEADS); // ord(-inf) ~ 0
        gat_edge_max<<<cdiv(ET, TB), TB, 0, stream>>>(ei, as_, ad_, mx);
        gat_edge_sum<<<cdiv(ET, TB), TB, 0, stream>>>(ei, as_, ad_, mx, dn);
        gat_edge_msg<<<cdiv((long)ET * HEADS, TB), TB, 0, stream>>>(ei, as_, ad_, mx, dn, bufH, agg);
        gat_post_elu_bf16<<<2048, TB, 0, stream>>>(agg, bias, out_bf);
    };

    // layer 1: input = bf16(x) [N,128]
    gat_layer(bufXb, W1t, IN_CH, a_src1, a_dst1, b1, bufXb);
    // layer 2: input = bf16(elu(gat1)) [N,192]
    gat_layer(bufXb, W2t, HC, a_src2, a_dst2, b2, bufXb);
    // fc1: [N,192] @ [192,64] + bf1 -> bf16  (no activation between fc layers)
    gemm(bufXb, Wf1t, bf1, nullptr, f1b, HC, F1);
    // fc2: [N,64] @ [64,32] + bf2 -> d_out (fp32)
    gemm(f1b, Wf2t, bf2, (float*)d_out, nullptr, F1, OUT_CH);
}
